// PLIPGNN_50946902065765
// MI455X (gfx1250) — matrix operand order repaired and checked by reading, not compiled, not run on hardware
//
#include <hip/hip_runtime.h>
#include <hip/hip_bf16.h>

typedef __attribute__((ext_vector_type(16))) _Float16 v16h;
typedef __attribute__((ext_vector_type(8)))  float    v8f;

#define N_NODES 100000
#define N_EDGES 1600000
#define IN_CH   41
#define ATOM_CH 21
#define EDGE_D  8
#define HID     64
#define NGRAPH  256
#define GN_EPS  1e-5f

// ---- WMMA fragment swizzle helpers (CDNA5 wave32, CDNA5 ISA §7.12.2) ----
// Fragments live in LDS already laid out per-lane: frag[lane] is one v16h, so a
// fragment read is a single 32B vector load (2x ds_load_b128).

// A matrix 16x32 f16: inverse map of {k = ((v>>2)<<4) + ((lane>>4)<<3) + ((v&3)<<1) + p}
__device__ __forceinline__ int a_lane(int k32, int m) {
  return (((k32 >> 3) & 1) << 4) | m;
}
__device__ __forceinline__ int a_elem(int k32) {
  const int v = ((k32 >> 4) << 2) | ((k32 & 7) >> 1);
  return (v << 1) | (k32 & 1);
}
// B matrix 32x16 f16: inverse map of {k = ((i>>3)<<4) + ((lane>>4)<<3) + (i&7)}
__device__ __forceinline__ int b_lane(int k32, int n16) {
  return (((k32 >> 3) & 1) << 4) | n16;
}
__device__ __forceinline__ int b_elem(int k32) { return ((k32 >> 4) << 3) | (k32 & 7); }

__device__ __forceinline__ void putA(_Float16* frag, int k32, int m, _Float16 val) {
  frag[a_lane(k32, m) * 16 + a_elem(k32)] = val;
}
__device__ __forceinline__ void putB(_Float16* frag, int k32, int n16, _Float16 val) {
  frag[b_lane(k32, n16) * 16 + b_elem(k32)] = val;
}

__device__ __forceinline__ v8f wmma_f16(v16h a, v16h b, v8f c) {
  return __builtin_amdgcn_wmma_f32_16x16x32_f16(false, a, false, b, (short)0, c, false, false);
}

// ---- 1) counts[g] = |{i : batch[i]==g}| ----
__global__ void counts_kernel(const int* __restrict__ batch, float* __restrict__ counts) {
  const int i = blockIdx.x * blockDim.x + threadIdx.x;
  if (i < N_NODES) atomicAdd(&counts[batch[i]], 1.0f);
}

// ---- 2) input encoder: dual 2-layer MLP + residue-mask select ----
__global__ __launch_bounds__(256) void encoder_kernel(
    const float* __restrict__ x,
    const float* __restrict__ ligW1, const float* __restrict__ ligb1,
    const float* __restrict__ ligW2, const float* __restrict__ ligb2,
    const float* __restrict__ prtW1, const float* __restrict__ prtb1,
    const float* __restrict__ prtW2, const float* __restrict__ prtb2,
    float* __restrict__ h) {
  // Wf[matrix][kblk][nt][lane] : pre-swizzled B fragments (32 KB)
  __shared__ v16h  Wf[4][2][4][32];
  __shared__ float Bl[4][64];
  // Ast[wave][kblk][lane] : pre-swizzled A fragments (16 KB)
  __shared__ v16h  Ast[8][2][32];
  __shared__ float rmask[8][16];

  const int tid  = threadIdx.x;
  const int wave = tid >> 5;
  const int lane = tid & 31;
  const int row0 = blockIdx.x * 128 + wave * 16;

  for (int idx = tid; idx < 64 * 64; idx += 256) {
    const int k = idx >> 6, n = idx & 63;
    const int kblk = k >> 5, k32 = k & 31, nt = n >> 4, n16 = n & 15;
    const float w1l = (k < IN_CH) ? ligW1[k * 64 + n] : 0.f;
    const float w1p = (k < IN_CH) ? prtW1[k * 64 + n] : 0.f;
    putB((_Float16*)&Wf[0][kblk][nt][0], k32, n16, (_Float16)w1l);
    putB((_Float16*)&Wf[1][kblk][nt][0], k32, n16, (_Float16)ligW2[idx]);
    putB((_Float16*)&Wf[2][kblk][nt][0], k32, n16, (_Float16)w1p);
    putB((_Float16*)&Wf[3][kblk][nt][0], k32, n16, (_Float16)prtW2[idx]);
  }
  for (int n = tid; n < 64; n += 256) {
    Bl[0][n] = ligb1[n]; Bl[1][n] = ligb2[n];
    Bl[2][n] = prtb1[n]; Bl[3][n] = prtb2[n];
  }
  // stage x tile (K padded 41->64) directly into A-fragment layout
  for (int t = lane; t < 16 * 64; t += 32) {
    const int r = t >> 6, c = t & 63;
    const int row = row0 + r;
    const float v = (row < N_NODES && c < IN_CH) ? x[row * IN_CH + c] : 0.f;
    putA((_Float16*)&Ast[wave][c >> 5][0], c & 31, r, (_Float16)v);
  }
  if (lane < 16) {
    const int row = row0 + lane;
    float s = 0.f;
    if (row < N_NODES)
      for (int c = ATOM_CH; c < IN_CH; ++c) s += fabsf(x[row * IN_CH + c]);
    rmask[wave][lane] = (s > 1e-6f) ? 1.f : 0.f;
  }
  __syncthreads();

  const v16h a0 = Ast[wave][0][lane];
  const v16h a1 = Ast[wave][1][lane];

  v8f res[2][4];
#pragma unroll
  for (int br = 0; br < 2; ++br) {
    v8f hid[4];
#pragma unroll
    for (int nt = 0; nt < 4; ++nt) {
      v8f c = {};
      c = wmma_f16(a0, Wf[br * 2][0][nt][lane], c);
      c = wmma_f16(a1, Wf[br * 2][1][nt][lane], c);
      hid[nt] = c;
    }
    __syncthreads();   // prior Ast reads done (all waves)
    {
      const int n = lane & 15, mb = (lane >> 4) << 3;
#pragma unroll
      for (int nt = 0; nt < 4; ++nt)
#pragma unroll
        for (int v = 0; v < 8; ++v) {
          const int col = nt * 16 + n;
          const float t = hid[nt][v] + Bl[br * 2][col];
          putA((_Float16*)&Ast[wave][col >> 5][0], col & 31, mb + v,
               (_Float16)fmaxf(t, 0.f));
        }
    }
    __syncthreads();   // hidden visible
    const v16h h0 = Ast[wave][0][lane];
    const v16h h1 = Ast[wave][1][lane];
#pragma unroll
    for (int nt = 0; nt < 4; ++nt) {
      v8f c = {};
      c = wmma_f16(h0, Wf[br * 2 + 1][0][nt][lane], c);
      c = wmma_f16(h1, Wf[br * 2 + 1][1][nt][lane], c);
      res[br][nt] = c;
    }
  }

  const int n = lane & 15, mb = (lane >> 4) << 3;
#pragma unroll
  for (int nt = 0; nt < 4; ++nt)
#pragma unroll
    for (int v = 0; v < 8; ++v) {
      const int m = mb + v, row = row0 + m;
      if (row < N_NODES) {
        const float lig = fmaxf(res[0][nt][v] + Bl[1][nt * 16 + n], 0.f);
        const float prt = fmaxf(res[1][nt][v] + Bl[3][nt * 16 + n], 0.f);
        h[row * 64 + nt * 16 + n] = (rmask[wave][m] > 0.5f) ? prt : lig;
      }
    }
}

// ---- 3) edge linear (WMMA) + gather h[src] + relu + scatter-add ----
__global__ __launch_bounds__(256) void edge_kernel(
    const float* __restrict__ edge_attr,
    const int* __restrict__ edge_index,
    const float* __restrict__ h,
    const float* __restrict__ We,   // [8][64] this layer
    const float* __restrict__ be,   // [64]
    float* __restrict__ aggr) {
  __shared__ v16h  Wf[4][32];      // K padded 8->32 (one kblk), 4 KB
  __shared__ float Bl[64];
  __shared__ v16h  Ast[8][32];     // per-wave edge tile, 8 KB

  const int tid  = threadIdx.x;
  const int wave = tid >> 5;
  const int lane = tid & 31;
  const int e0   = blockIdx.x * 128 + wave * 16;

  for (int idx = tid; idx < 32 * 64; idx += 256) {
    const int k = idx >> 6, n = idx & 63;
    const float w = (k < EDGE_D) ? We[k * 64 + n] : 0.f;
    putB((_Float16*)&Wf[n >> 4][0], k, n & 15, (_Float16)w);
  }
  for (int n = tid; n < 64; n += 256) Bl[n] = be[n];

  for (int t = lane; t < 16 * 32; t += 32) {
    const int r = t >> 5, c = t & 31;
    const int e = e0 + r;
    const float v = (e < N_EDGES && c < EDGE_D) ? edge_attr[e * EDGE_D + c] : 0.f;
    putA((_Float16*)&Ast[wave][0], c, r, (_Float16)v);
  }
  __syncthreads();

  const v16h a = Ast[wave][lane];
  const int n = lane & 15, mb = (lane >> 4) << 3;

  int sv[8], dv[8];
#pragma unroll
  for (int v = 0; v < 8; ++v) {
    const int e = e0 + mb + v;
    sv[v] = (e < N_EDGES) ? edge_index[e] : 0;
    dv[v] = (e < N_EDGES) ? edge_index[N_EDGES + e] : 0;
  }

#pragma unroll
  for (int nt = 0; nt < 4; ++nt) {
    v8f c = {};
    c = wmma_f16(a, Wf[nt][lane], c);
    const int col = nt * 16 + n;
#pragma unroll
    for (int v = 0; v < 8; ++v) {
      const int e = e0 + mb + v;
      if (e < N_EDGES) {
        float msg = c[v] + Bl[col] + h[(size_t)sv[v] * 64 + col];
        msg = fmaxf(msg, 0.f);
        atomicAdd(&aggr[(size_t)dv[v] * 64 + col], msg);
      }
    }
  }
}

// ---- 4) node MLP: buf := relu((h+aggr)@W1+b1)@W2+b2 ; sums[g] += buf ----
__global__ __launch_bounds__(256) void node_mlp_kernel(
    const float* __restrict__ h,
    float* __restrict__ buf,          // in: aggr, out: hmlp (in place)
    const float* __restrict__ W1, const float* __restrict__ b1,
    const float* __restrict__ W2, const float* __restrict__ b2,
    const int* __restrict__ batch,
    float* __restrict__ sums) {
  __shared__ v16h  Wf[2][2][4][32];   // 16 KB
  __shared__ float Bl[2][64];
  __shared__ v16h  Ast[8][2][32];     // 16 KB

  const int tid  = threadIdx.x;
  const int wave = tid >> 5;
  const int lane = tid & 31;
  const int row0 = blockIdx.x * 128 + wave * 16;

  for (int idx = tid; idx < 64 * 64; idx += 256) {
    const int k = idx >> 6, n = idx & 63;
    const int kblk = k >> 5, k32 = k & 31, nt = n >> 4, n16 = n & 15;
    putB((_Float16*)&Wf[0][kblk][nt][0], k32, n16, (_Float16)W1[idx]);
    putB((_Float16*)&Wf[1][kblk][nt][0], k32, n16, (_Float16)W2[idx]);
  }
  for (int n = tid; n < 64; n += 256) { Bl[0][n] = b1[n]; Bl[1][n] = b2[n]; }

  for (int t = lane; t < 16 * 64; t += 32) {
    const int r = t >> 6, c = t & 63;
    const int row = row0 + r;
    const float z = (row < N_NODES)
        ? (h[(size_t)row * 64 + c] + buf[(size_t)row * 64 + c]) : 0.f;
    putA((_Float16*)&Ast[wave][c >> 5][0], c & 31, r, (_Float16)z);
  }
  __syncthreads();

  const v16h a0 = Ast[wave][0][lane];
  const v16h a1 = Ast[wave][1][lane];
  v8f hid[4];
#pragma unroll
  for (int nt = 0; nt < 4; ++nt) {
    v8f c = {};
    c = wmma_f16(a0, Wf[0][0][nt][lane], c);
    c = wmma_f16(a1, Wf[0][1][nt][lane], c);
    hid[nt] = c;
  }
  __syncthreads();
  const int n = lane & 15, mb = (lane >> 4) << 3;
#pragma unroll
  for (int nt = 0; nt < 4; ++nt)
#pragma unroll
    for (int v = 0; v < 8; ++v) {
      const int col = nt * 16 + n;
      const float t = hid[nt][v] + Bl[0][col];
      putA((_Float16*)&Ast[wave][col >> 5][0], col & 31, mb + v,
           (_Float16)fmaxf(t, 0.f));
    }
  __syncthreads();
  const v16h h0 = Ast[wave][0][lane];
  const v16h h1 = Ast[wave][1][lane];

  int gv[8]; bool ok[8];
#pragma unroll
  for (int v = 0; v < 8; ++v) {
    const int row = row0 + mb + v;
    ok[v] = row < N_NODES;
    gv[v] = ok[v] ? batch[row] : 0;
  }
#pragma unroll
  for (int nt = 0; nt < 4; ++nt) {
    v8f c = {};
    c = wmma_f16(h0, Wf[1][0][nt][lane], c);
    c = wmma_f16(h1, Wf[1][1][nt][lane], c);
    const int col = nt * 16 + n;
#pragma unroll
    for (int v = 0; v < 8; ++v)
      if (ok[v]) {
        const int row = row0 + mb + v;
        const float val = c[v] + Bl[1][col];
        buf[(size_t)row * 64 + col] = val;
        atomicAdd(&sums[gv[v] * 64 + col], val);
      }
  }
}

// ---- 5) GraphNorm pass A: out = hmlp - a*mean ; var[g] += out^2 ----
__global__ void gnorm_var_kernel(float* __restrict__ buf,
    const int* __restrict__ batch, const float* __restrict__ sums,
    const float* __restrict__ counts, const float* __restrict__ alpha,
    float* __restrict__ var) {
  const int i = blockIdx.x * blockDim.x + threadIdx.x;
  if (i >= N_NODES * 64) return;
  const int row = i >> 6, c = i & 63;
  const int g = batch[row];
  const float cnt = fmaxf(counts[g], 1.f);
  const float mean = sums[g * 64 + c] / cnt;
  const float out = buf[i] - alpha[c] * mean;
  buf[i] = out;
  atomicAdd(&var[g * 64 + c], out * out);
}

// ---- 6) GraphNorm pass B: h = relu(w*out*rsqrt(var+eps)+b) ; opt. pool ----
__global__ void gnorm_fin_kernel(const float* __restrict__ buf,
    const int* __restrict__ batch, const float* __restrict__ var,
    const float* __restrict__ counts, const float* __restrict__ w,
    const float* __restrict__ b, float* __restrict__ h,
    float* __restrict__ gpool, int do_pool) {
  const int i = blockIdx.x * blockDim.x + threadIdx.x;
  if (i >= N_NODES * 64) return;
  const int row = i >> 6, c = i & 63;
  const int g = batch[row];
  const float cnt = fmaxf(counts[g], 1.f);
  const float vr = var[g * 64 + c] / cnt;
  const float val = fmaxf(w[c] * (buf[i] * rsqrtf(vr + GN_EPS)) + b[c], 0.f);
  h[i] = val;
  if (do_pool) atomicAdd(&gpool[g * 64 + c], val);
}

// ---- 7) readout: out[g] = relu(gpool@W1+b1)@W2+b2 (G=256, tiny) ----
__global__ __launch_bounds__(256) void readout_kernel(
    const float* __restrict__ gpool,
    const float* __restrict__ W1, const float* __restrict__ b1,
    const float* __restrict__ W2, const float* __restrict__ b2,
    float* __restrict__ out) {
  __shared__ float W1s[64 * 64];
  const int tid = threadIdx.x;
  for (int idx = tid; idx < 64 * 64; idx += 256) W1s[idx] = W1[idx];
  __syncthreads();
  float row[64];
  for (int k = 0; k < 64; ++k) row[k] = gpool[tid * 64 + k];
  float acc = b2[0];
  for (int n = 0; n < 64; ++n) {
    float t = b1[n];
    for (int k = 0; k < 64; ++k) t += row[k] * W1s[k * 64 + n];
    acc += fmaxf(t, 0.f) * W2[n];
  }
  out[tid] = acc;
}

extern "C" void kernel_launch(void* const* d_in, const int* in_sizes, int n_in,
                              void* d_out, int out_size, void* d_ws, size_t ws_size,
                              hipStream_t stream) {
  const float* x      = (const float*)d_in[0];
  const int*   eidx   = (const int*)  d_in[1];
  const float* eattr  = (const float*)d_in[2];
  const int*   batch  = (const int*)  d_in[3];
  const float* ligW1  = (const float*)d_in[4];
  const float* ligb1  = (const float*)d_in[5];
  const float* ligW2  = (const float*)d_in[6];
  const float* ligb2  = (const float*)d_in[7];
  const float* prtW1  = (const float*)d_in[8];
  const float* prtb1  = (const float*)d_in[9];
  const float* prtW2  = (const float*)d_in[10];
  const float* prtb2  = (const float*)d_in[11];
  const float* convWe = (const float*)d_in[12];
  const float* convbe = (const float*)d_in[13];
  const float* convW1 = (const float*)d_in[14];
  const float* convb1 = (const float*)d_in[15];
  const float* convW2 = (const float*)d_in[16];
  const float* convb2 = (const float*)d_in[17];
  const float* norm_w = (const float*)d_in[18];
  const float* norm_b = (const float*)d_in[19];
  const float* norm_a = (const float*)d_in[20];
  const float* outW1  = (const float*)d_in[21];
  const float* outb1  = (const float*)d_in[22];
  const float* outW2  = (const float*)d_in[23];
  const float* outb2  = (const float*)d_in[24];

  // workspace layout (~51.4 MB)
  const size_t HN = (size_t)N_NODES * 64;
  float* h      = (float*)d_ws;
  float* buf    = h + HN;                 // aggr -> hmlp -> out (in place)
  float* sums   = buf + HN;
  float* var    = sums + NGRAPH * 64;
  float* counts = var + NGRAPH * 64;
  float* gpool  = counts + NGRAPH;

  hipMemsetAsync(counts, 0, NGRAPH * sizeof(float), stream);
  hipMemsetAsync(gpool,  0, NGRAPH * 64 * sizeof(float), stream);
  counts_kernel<<<(N_NODES + 255) / 256, 256, 0, stream>>>(batch, counts);

  const int node_blocks = (N_NODES + 127) / 128;
  const int edge_blocks = (N_EDGES + 127) / 128;
  const int elem_blocks = (N_NODES * 64 + 255) / 256;

  encoder_kernel<<<node_blocks, 256, 0, stream>>>(
      x, ligW1, ligb1, ligW2, ligb2, prtW1, prtb1, prtW2, prtb2, h);

  for (int l = 0; l < 3; ++l) {
    hipMemsetAsync(buf, 0, HN * sizeof(float), stream);
    edge_kernel<<<edge_blocks, 256, 0, stream>>>(
        eattr, eidx, h, convWe + (size_t)l * EDGE_D * 64, convbe + l * 64, buf);

    hipMemsetAsync(sums, 0, NGRAPH * 64 * sizeof(float), stream);
    node_mlp_kernel<<<node_blocks, 256, 0, stream>>>(
        h, buf, convW1 + (size_t)l * 64 * 64, convb1 + l * 64,
        convW2 + (size_t)l * 64 * 64, convb2 + l * 64, batch, sums);

    hipMemsetAsync(var, 0, NGRAPH * 64 * sizeof(float), stream);
    gnorm_var_kernel<<<elem_blocks, 256, 0, stream>>>(
        buf, batch, sums, counts, norm_a + l * 64, var);

    gnorm_fin_kernel<<<elem_blocks, 256, 0, stream>>>(
        buf, batch, var, counts, norm_w + l * 64, norm_b + l * 64,
        h, gpool, (l == 2) ? 1 : 0);
  }

  readout_kernel<<<1, 256, 0, stream>>>(gpool, outW1, outb1, outW2, outb2,
                                        (float*)d_out);
}
